// DeepFM_43757126812202
// MI455X (gfx1250) — matrix operand order repaired and checked by reading, not compiled, not run on hardware
//
#include <hip/hip_runtime.h>
#include <hip/hip_bf16.h>
#include <math.h>

// ---------------------------------------------------------------------------
// DeepFM inference for MI455X (gfx1250, wave32, WMMA bf16 16x16x32)
//   - GEMMs on v_wmma_f32_16x16x32_bf16, 4 N-tiles per wave (A reuse x4)
//   - A tiles staged in LDS via global_load_async_to_lds_b128 (ASYNCcnt path)
// ---------------------------------------------------------------------------

typedef __attribute__((ext_vector_type(16))) __bf16 v16bf;
typedef __attribute__((ext_vector_type(8)))  float  v8f;
typedef __attribute__((ext_vector_type(4)))  int    v4i;

#define B_ROWS 16384
#define F_FEAT 26
#define V_VOC  100000
#define D_DIM  16
#define K1_DIM 416   // F*D = 13*32
#define H1_DIM 256
#define H2_DIM 128

// ---- workspace layout (bytes, 256-aligned sections) ----
#define OFF_W1F   ((size_t)0)                         // 416*256 bf16 = 212992
#define OFF_W2F   ((size_t)212992)                    // 256*128 bf16 = 65536
#define OFF_SC1   ((size_t)278528)                    // 256 f32
#define OFF_BI1   ((size_t)279552)                    // 256 f32
#define OFF_SC2   ((size_t)280576)                    // 128 f32
#define OFF_BI2   ((size_t)281088)                    // 128 f32
#define OFF_FM    ((size_t)281600)                    // 16384 f32 = 65536
#define OFF_A1    ((size_t)347136)                    // 16384*416 bf16 = 13631488
#define OFF_H1    ((size_t)13978624)                  // 16384*256 bf16 = 8388608
#define OFF_H2    ((size_t)22367232)                  // 16384*128 bf16 = 4194304

// ---- async global->LDS copy (gfx1250), guarded by toolchain support ----
#if defined(__has_builtin)
#if __has_builtin(__builtin_amdgcn_global_load_async_to_lds_b128)
#define HAVE_ASYNC_LDS 1
#endif
#endif

#if defined(HAVE_ASYNC_LDS)
#define AS1 __attribute__((address_space(1)))
#define AS3 __attribute__((address_space(3)))
__device__ __forceinline__ void async_copy_b128(const void* g, void* l) {
    // signature: (int4 AS1* src, int4 AS3* dst, imm offset, imm cpol)
    __builtin_amdgcn_global_load_async_to_lds_b128((AS1 v4i*)g, (AS3 v4i*)l,
                                                   /*offset=*/0, /*cpol=*/0);
}
#endif

// ---------------------------------------------------------------------------
// Fold BatchNorm (eval) + bias into per-column scale/bias:
//   bn(x@W + b) = (x@W)*s + ((b - m)*s + beta),  s = g * rsqrt(v + eps)
// ---------------------------------------------------------------------------
__global__ void prep_bn(const float* __restrict__ b1, const float* __restrict__ g1,
                        const float* __restrict__ be1, const float* __restrict__ m1,
                        const float* __restrict__ v1,
                        const float* __restrict__ b2, const float* __restrict__ g2,
                        const float* __restrict__ be2, const float* __restrict__ m2,
                        const float* __restrict__ v2,
                        float* __restrict__ sc1, float* __restrict__ bi1,
                        float* __restrict__ sc2, float* __restrict__ bi2) {
    int i = threadIdx.x;
    if (i < H1_DIM) {
        float s = g1[i] * rsqrtf(v1[i] + 1e-5f);
        sc1[i] = s;
        bi1[i] = (b1[i] - m1[i]) * s + be1[i];
    } else if (i < H1_DIM + H2_DIM) {
        int j = i - H1_DIM;
        float s = g2[j] * rsqrtf(v2[j] + 1e-5f);
        sc2[j] = s;
        bi2[j] = (b2[j] - m2[j]) * s + be2[j];
    }
}

// ---------------------------------------------------------------------------
// Pre-swizzle W [K,N] (row-major f32) into bf16 B-fragment layout.
// Per 32x16 K-tile x N-tile: lane L holds column n = nt*16 + (L&15),
// K-range base = kt*32 + (L>>4)*16, 16 contiguous bf16 (32B) per lane.
// ---------------------------------------------------------------------------
__global__ void prep_wfrag(const float* __restrict__ W, __bf16* __restrict__ dst,
                           int ktiles, int ntiles, int N) {
    int t = blockIdx.x * blockDim.x + threadIdx.x;
    int total = ktiles * ntiles * 32;
    if (t >= total) return;
    int lane  = t & 31;
    int tile  = t >> 5;
    int nt    = tile % ntiles;
    int kt    = tile / ntiles;
    int n     = nt * 16 + (lane & 15);
    int kbase = kt * 32 + (lane >> 4) * 16;
    __bf16* out = dst + ((size_t)(tile * 32 + lane)) * 16;
#pragma unroll
    for (int e = 0; e < 16; ++e)
        out[e] = (__bf16)W[(size_t)(kbase + e) * N + n];
}

// ---------------------------------------------------------------------------
// Embedding gather + FM terms. 16 threads per row (thread = dim d).
// Writes A1 bf16 [B,416] row-major; FM[b] = fm1 + 0.5*sum_d(s_d^2 - ss_d).
// ---------------------------------------------------------------------------
__global__ __launch_bounds__(256)
void gather_fm(const int* __restrict__ X, const float* __restrict__ emb1,
               const float* __restrict__ emb2, __bf16* __restrict__ A1,
               float* __restrict__ FM) {
    int b = blockIdx.x * 16 + (threadIdx.x >> 4);
    int d = threadIdx.x & 15;
    const int* xr = X + (size_t)b * F_FEAT;
    __bf16* arow  = A1 + (size_t)b * K1_DIM;
    float s = 0.f, ss = 0.f, fm1 = 0.f;
#pragma unroll
    for (int f = 0; f < F_FEAT; ++f) {
        int idx = xr[f];
        float e = emb2[((size_t)f * V_VOC + (size_t)idx) * D_DIM + d];
        s += e;
        ss += e * e;
        arow[f * D_DIM + d] = (__bf16)e;
        if ((f & 15) == d)  // each feature's e1 handled by exactly one lane
            fm1 += emb1[(size_t)f * V_VOC + (size_t)idx];
    }
    float part = fm1 + 0.5f * (s * s - ss);
    part += __shfl_xor(part, 8, 16);
    part += __shfl_xor(part, 4, 16);
    part += __shfl_xor(part, 2, 16);
    part += __shfl_xor(part, 1, 16);
    if (d == 0) FM[b] = part;
}

// ---------------------------------------------------------------------------
// GEMM (bf16 WMMA, f32 acc) + fused scale/bias + ReLU, bf16 output.
// Each wave computes a 16x64 strip (4 N-tiles): one A fragment feeds 4 WMMAs.
// The block stages its A tiles (MTB x 16 rows x K) in LDS once,
// via async global->LDS copies when available.
// ---------------------------------------------------------------------------
template <int KT, int NT>
__global__ __launch_bounds__(256)
void gemm_bn_relu(const __bf16* __restrict__ A, const __bf16* __restrict__ Bfrag,
                  const float* __restrict__ scale, const float* __restrict__ bias,
                  __bf16* __restrict__ Out) {
    constexpr int K   = KT * 32;
    constexpr int N   = NT * 16;
    constexpr int NG  = 4;            // N-tiles per wave
    constexpr int WPM = NT / NG;      // waves per M-tile
    constexpr int MTB = 8 / WPM;      // M-tiles per block (8 waves/block)

    __shared__ alignas(16) __bf16 Alds[MTB * 16 * K];

    int lane = threadIdx.x & 31;
    int wave = threadIdx.x >> 5;
    int mloc = wave / WPM;            // which staged M-tile this wave uses
    int ng   = wave % WPM;            // which group of 4 N-tiles
    int row0 = (blockIdx.x * MTB + mloc) * 16;

    // ---- cooperative stage of A tiles into LDS (contiguous rows) ----
    {
        constexpr int NCHUNK = MTB * 16 * K * 2 / 16;  // 16B chunks
        const __bf16* gbase = A + (size_t)blockIdx.x * MTB * 16 * K;
        for (int i = threadIdx.x; i < NCHUNK; i += 256) {
#if defined(HAVE_ASYNC_LDS)
            async_copy_b128(gbase + i * 8, &Alds[i * 8]);
#else
            *(uint4*)(&Alds[i * 8]) = *(const uint4*)(gbase + i * 8);
#endif
        }
#if defined(HAVE_ASYNC_LDS)
        asm volatile("s_wait_asynccnt 0x0" ::: "memory");
#endif
        __syncthreads();
    }

    int m_a   = lane & 15;   // A row within tile
    int khalf = lane >> 4;   // which K-half this lane holds

    v8f acc[NG] = {};
    // A (LDS): elems 0..15 -> K = kt*32 + khalf*16 + e.
    const __bf16* arow = &Alds[(mloc * 16 + m_a) * K] + khalf * 16;

#pragma unroll
    for (int kt = 0; kt < KT; ++kt) {
        union { v16bf v; uint4 q[2]; } a;
        a.q[0] = *(const uint4*)(arow + kt * 32);
        a.q[1] = *(const uint4*)(arow + kt * 32 + 8);
#pragma unroll
        for (int j = 0; j < NG; ++j) {
            int nt = ng * NG + j;
            union { v16bf v; uint4 q[2]; } bm;
            const __bf16* bp = Bfrag + (((size_t)kt * NT + nt) * 32 + lane) * 16;
            bm.q[0] = *(const uint4*)(bp);
            bm.q[1] = *(const uint4*)(bp + 8);
            acc[j] = __builtin_amdgcn_wmma_f32_16x16x32_bf16(
                /*neg_a=*/false, a.v, /*neg_b=*/false, bm.v,
                /*c_mod=*/(short)0, acc[j], /*reuse_a=*/false, /*reuse_b=*/false);
        }
    }

    // C/D layout: VGPR r -> M = r + 8*khalf, N = lane&15.
    int nbase = ng * NG * 16 + (lane & 15);
#pragma unroll
    for (int j = 0; j < NG; ++j) {
        int n = nbase + j * 16;
        float sc = scale[n], bi = bias[n];
#pragma unroll
        for (int r = 0; r < 8; ++r) {
            int m = row0 + khalf * 8 + r;
            float v = acc[j][r] * sc + bi;
            v = v > 0.f ? v : 0.f;
            Out[(size_t)m * N + n] = (__bf16)v;
        }
    }
}

// ---------------------------------------------------------------------------
// Final: logits = h2 @ W3 + b3 + FM -> sigmoid. One row per wave.
// ---------------------------------------------------------------------------
__global__ __launch_bounds__(256)
void final_k(const __bf16* __restrict__ H2, const float* __restrict__ W3,
             const float* __restrict__ b3, const float* __restrict__ FM,
             float* __restrict__ out) {
    int lane = threadIdx.x & 31;
    int b = blockIdx.x * 8 + (threadIdx.x >> 5);
    const __bf16* h = H2 + (size_t)b * H2_DIM;
    float sum = 0.f;
#pragma unroll
    for (int j = 0; j < 4; ++j) {
        int i = lane + 32 * j;
        sum += (float)h[i] * W3[i];
    }
    sum += __shfl_xor(sum, 16);
    sum += __shfl_xor(sum, 8);
    sum += __shfl_xor(sum, 4);
    sum += __shfl_xor(sum, 2);
    sum += __shfl_xor(sum, 1);
    if (lane == 0) {
        float logit = sum + b3[0] + FM[b];
        out[b] = 1.f / (1.f + __expf(-logit));
    }
}

// ---------------------------------------------------------------------------
extern "C" void kernel_launch(void* const* d_in, const int* in_sizes, int n_in,
                              void* d_out, int out_size, void* d_ws, size_t ws_size,
                              hipStream_t stream) {
    const int*   X    = (const int*)d_in[0];
    const float* emb1 = (const float*)d_in[1];
    const float* emb2 = (const float*)d_in[2];
    const float* W1   = (const float*)d_in[3];
    const float* b1   = (const float*)d_in[4];
    const float* g1   = (const float*)d_in[5];
    const float* be1  = (const float*)d_in[6];
    const float* m1   = (const float*)d_in[7];
    const float* v1   = (const float*)d_in[8];
    const float* W2   = (const float*)d_in[9];
    const float* b2   = (const float*)d_in[10];
    const float* g2   = (const float*)d_in[11];
    const float* be2  = (const float*)d_in[12];
    const float* m2   = (const float*)d_in[13];
    const float* v2   = (const float*)d_in[14];
    const float* W3   = (const float*)d_in[15];
    const float* b3   = (const float*)d_in[16];
    float* out = (float*)d_out;

    unsigned char* ws = (unsigned char*)d_ws;
    __bf16* W1f = (__bf16*)(ws + OFF_W1F);
    __bf16* W2f = (__bf16*)(ws + OFF_W2F);
    float*  sc1 = (float*)(ws + OFF_SC1);
    float*  bi1 = (float*)(ws + OFF_BI1);
    float*  sc2 = (float*)(ws + OFF_SC2);
    float*  bi2 = (float*)(ws + OFF_BI2);
    float*  FM  = (float*)(ws + OFF_FM);
    __bf16* A1  = (__bf16*)(ws + OFF_A1);
    __bf16* H1  = (__bf16*)(ws + OFF_H1);
    __bf16* H2  = (__bf16*)(ws + OFF_H2);

    // 1) fold BN params
    prep_bn<<<1, 384, 0, stream>>>(b1, g1, be1, m1, v1, b2, g2, be2, m2, v2,
                                   sc1, bi1, sc2, bi2);
    // 2) swizzle weights into bf16 WMMA B-fragments
    prep_wfrag<<<26, 256, 0, stream>>>(W1, W1f, 13, 16, H1_DIM);  // 13*16*32 lanes
    prep_wfrag<<<8, 256, 0, stream>>>(W2, W2f, 8, 8, H2_DIM);     // 8*8*32 lanes
    // 3) embedding gather + FM terms
    gather_fm<<<B_ROWS / 16, 256, 0, stream>>>(X, emb1, emb2, A1, FM);
    // 4) layer 1: [16384,416] x [416,256]; 2 M-tiles/block -> 512 blocks
    gemm_bn_relu<13, 16><<<(B_ROWS / 16) / 2, 256, 0, stream>>>(A1, W1f, sc1, bi1, H1);
    // 5) layer 2: [16384,256] x [256,128]; 4 M-tiles/block -> 256 blocks
    gemm_bn_relu<8, 8><<<(B_ROWS / 16) / 4, 256, 0, stream>>>(H1, W2f, sc2, bi2, H2);
    // 6) final dot + FM + sigmoid
    final_k<<<B_ROWS / 8, 256, 0, stream>>>(H2, W3, b3, FM, out);
}
